// BoltzmannMoE_54503134986829
// MI455X (gfx1250) — compile-verified, matrix operand order checked by reading, not run on hardware
//
#include <hip/hip_runtime.h>

#define NTOK 8192
#define DIM  1024
#define HDIM 4096
#define ODIM 1024
#define NEXP 8
#define INV_TEMP 0.36787944117144233f  // 1/2.718281828459045

typedef __attribute__((ext_vector_type(16))) __bf16 v16bf;
typedef __attribute__((ext_vector_type(8)))  __bf16 v8bf;
typedef __attribute__((ext_vector_type(8)))  float  v8f;

// Tiling for the fused expert FFN kernel
#define BM 32          // tokens per block
#define BO 512         // output columns per block (grid.y = ODIM/BO)
#define HC 64          // hidden chunk
#define KB 128         // K-blocking for GEMM1 (over DIM)
#define XS_STRIDE 1032 // bf16 elems per x row in LDS (DIM + 8 pad)
#define W1_STRIDE 136  // k-stride of transposed W1 panel (KB + 8)
#define HS_STRIDE 72   // h-stride of h-chunk rows (HC + 8)
#define W2_STRIDE 72   // h-stride of transposed W2 panel rows (HC + 8)

// ---------------------------------------------------------------------------
// Gating: one wave32 per token. scores = (x@Wg + bg)/T, softmax, top-2,
// renormalize, append (token, weight) to the chosen experts' lists.
// ---------------------------------------------------------------------------
__global__ __launch_bounds__(256)
void moe_gate(const float* __restrict__ x, const float* __restrict__ Wg,
              const float* __restrict__ bg, int* __restrict__ counts,
              int* __restrict__ tok_list, float* __restrict__ w_list) {
  const int token = blockIdx.x * 8 + (threadIdx.x >> 5);
  const int lane  = threadIdx.x & 31;
  if (token >= NTOK) return;

  float p[NEXP];
  #pragma unroll
  for (int e = 0; e < NEXP; ++e) p[e] = 0.0f;

  const float* xr = x + (size_t)token * DIM;
  for (int d = lane; d < DIM; d += 32) {
    const float xv = xr[d];
    const float* wr = Wg + (size_t)d * NEXP;
    #pragma unroll
    for (int e = 0; e < NEXP; ++e) p[e] += xv * wr[e];
  }
  #pragma unroll
  for (int off = 16; off >= 1; off >>= 1) {
    #pragma unroll
    for (int e = 0; e < NEXP; ++e) p[e] += __shfl_xor(p[e], off, 32);
  }

  float sc[NEXP], mx = -1e30f;
  #pragma unroll
  for (int e = 0; e < NEXP; ++e) {
    sc[e] = (p[e] + bg[e]) * INV_TEMP;
    mx = fmaxf(mx, sc[e]);
  }
  float se = 0.0f;
  #pragma unroll
  for (int e = 0; e < NEXP; ++e) { sc[e] = __expf(sc[e] - mx); se += sc[e]; }

  int i0 = 0;
  #pragma unroll
  for (int e = 1; e < NEXP; ++e) if (sc[e] > sc[i0]) i0 = e;
  int i1 = (i0 == 0) ? 1 : 0;
  #pragma unroll
  for (int e = 0; e < NEXP; ++e) if (e != i0 && sc[e] > sc[i1]) i1 = e;

  const float v0 = sc[i0] / se, v1 = sc[i1] / se;
  const float inv = 1.0f / (v0 + v1 + 1e-8f);
  const float w0 = v0 * inv, w1 = v1 * inv;

  if (lane == 0) {
    int s0 = atomicAdd(&counts[i0], 1);
    tok_list[i0 * NTOK + s0] = token; w_list[i0 * NTOK + s0] = w0;
    int s1 = atomicAdd(&counts[i1], 1);
    tok_list[i1 * NTOK + s1] = token; w_list[i1 * NTOK + s1] = w1;
  }
}

// ---------------------------------------------------------------------------
// Fused routed expert FFN: out[tok] += w * (relu(x@W1+b1)@W2 + b2), all
// matmuls via v_wmma_f32_16x16x32_bf16 with f32 accumulation.
// ---------------------------------------------------------------------------
__global__ __launch_bounds__(256)
void moe_expert_ffn(const float* __restrict__ x,
                    const float* __restrict__ W1, const float* __restrict__ b1,
                    const float* __restrict__ W2, const float* __restrict__ b2,
                    const int* __restrict__ counts,
                    const int* __restrict__ tok_list,
                    const float* __restrict__ w_list,
                    float* __restrict__ out) {
  const int e   = blockIdx.z;
  const int cnt = counts[e];
  const int mt  = blockIdx.x;
  if (mt * BM >= cnt) return;
  const int obase = blockIdx.y * BO;

  extern __shared__ char smem_raw[];
  __bf16* xs   = (__bf16*)smem_raw;            // [BM][XS_STRIDE]
  __bf16* w1s  = xs  + BM * XS_STRIDE;         // [HC][W1_STRIDE] (h-major, k contig)
  __bf16* hs   = w1s + HC * W1_STRIDE;         // [BM][HS_STRIDE]
  __bf16* w2s  = hs  + BM * HS_STRIDE;         // [BO][W2_STRIDE] (o-major, h contig)
  int*    toks = (int*)(w2s + BO * W2_STRIDE); // [BM]
  float*  wts  = (float*)(toks + BM);          // [BM]

  const int tid   = threadIdx.x;
  const int wave  = tid >> 5;
  const int lane  = tid & 31;
  const int l16   = lane & 15;
  const int kbase = (lane < 16) ? 0 : 8;   // per-half-wave K base (16-bit A/B layout)
  const int mrow  = (wave >> 2) * 16;      // wave's M half: 0 or 16
  const int wsub  = wave & 3;              // wave's h-subtile (GEMM1) / o-block (GEMM2)

  const float* W1e = W1 + (size_t)e * DIM * HDIM;
  const float* W2e = W2 + (size_t)e * HDIM * ODIM;

  if (tid < BM) {
    const int row = mt * BM + tid;
    toks[tid] = (row < cnt) ? tok_list[e * NTOK + row] : 0;
    wts[tid]  = (row < cnt) ? w_list[e * NTOK + row] : 0.0f;
  }
  __syncthreads();

  // Stage x tile fp32 -> bf16 (coalesced float4 reads)
  for (int i = tid; i < BM * (DIM / 4); i += 256) {
    const int r  = i >> 8;          // DIM/4 == 256
    const int c4 = (i & 255) << 2;
    const float4 v = *(const float4*)(x + (size_t)toks[r] * DIM + c4);
    __bf16* dst = xs + r * XS_STRIDE + c4;
    dst[0] = (__bf16)v.x; dst[1] = (__bf16)v.y;
    dst[2] = (__bf16)v.z; dst[3] = (__bf16)v.w;
  }

  v8f acc[8];
  #pragma unroll
  for (int t = 0; t < 8; ++t) acc[t] = (v8f){};

  for (int hc = 0; hc < HDIM; hc += HC) {
    // ---- GEMM1: h[BM][HC] = x[BM][DIM] @ W1[DIM][HC], K-blocked by KB ----
    v8f acc1 = (v8f){};
    for (int kb = 0; kb < DIM; kb += KB) {
      __syncthreads();
      // stage W1 panel transposed: w1s[h][k] = W1e[kb+k][hc+h]
      for (int i = tid; i < HC * KB; i += 256) {
        const int k = i >> 6;  // /HC
        const int h = i & 63;
        w1s[h * W1_STRIDE + k] = (__bf16)W1e[(size_t)(kb + k) * HDIM + hc + h];
      }
      __syncthreads();
      #pragma unroll
      for (int kk = 0; kk < KB; kk += 32) {
        union { v16bf v; v8bf h[2]; } a, b;
        const __bf16* ap = xs + (mrow + l16) * XS_STRIDE + kb + kk + kbase;
        a.h[0] = *(const v8bf*)(ap);
        a.h[1] = *(const v8bf*)(ap + 16);
        const __bf16* bp = w1s + (wsub * 16 + l16) * W1_STRIDE + kk + kbase;
        b.h[0] = *(const v8bf*)(bp);
        b.h[1] = *(const v8bf*)(bp + 16);
        acc1 = __builtin_amdgcn_wmma_f32_16x16x32_bf16(false, a.v, false, b.v,
                                                       (short)0, acc1, false, false);
      }
    }
    // relu + b1 -> bf16 h-chunk in LDS (C layout: VGPR r -> M=r / M=8+r)
    {
      const int hloc = wsub * 16 + l16;
      const float bv = b1[e * HDIM + hc + hloc];
      const int mhi  = (lane >> 4) << 3;
      #pragma unroll
      for (int r = 0; r < 8; ++r) {
        const float hv = fmaxf(acc1[r] + bv, 0.0f);
        hs[(mrow + r + mhi) * HS_STRIDE + hloc] = (__bf16)hv;
      }
    }
    // stage W2 panel transposed: w2s[o][h] = W2e[hc+h][obase+o]
    for (int i = tid; i < HC * BO; i += 256) {
      const int h = i >> 9;  // /BO
      const int o = i & 511;
      w2s[o * W2_STRIDE + h] = (__bf16)W2e[(size_t)(hc + h) * ODIM + obase + o];
    }
    if (hc + HC < HDIM) {  // prefetch a slice of the next W2 panel
      const int h = tid >> 2, o = (tid & 3) << 7;
      __builtin_prefetch(&W2e[(size_t)(hc + HC + h) * ODIM + obase + o], 0, 1);
    }
    __syncthreads();
    // ---- GEMM2: acc += h_chunk[BM][HC] @ W2[HC][BO] ----
    #pragma unroll
    for (int t = 0; t < 8; ++t) {
      #pragma unroll
      for (int kk = 0; kk < HC; kk += 32) {
        union { v16bf v; v8bf h[2]; } a, b;
        const __bf16* ap = hs + (mrow + l16) * HS_STRIDE + kk + kbase;
        a.h[0] = *(const v8bf*)(ap);
        a.h[1] = *(const v8bf*)(ap + 16);
        const __bf16* bp = w2s + (wsub * 128 + t * 16 + l16) * W2_STRIDE + kk + kbase;
        b.h[0] = *(const v8bf*)(bp);
        b.h[1] = *(const v8bf*)(bp + 16);
        acc[t] = __builtin_amdgcn_wmma_f32_16x16x32_bf16(false, a.v, false, b.v,
                                                         (short)0, acc[t], false, false);
      }
    }
  }

  // Epilogue: out[tok] += w * (acc + b2)
  {
    const int mhi = (lane >> 4) << 3;
    #pragma unroll
    for (int t = 0; t < 8; ++t) {
      const int ocol = obase + wsub * 128 + t * 16 + l16;
      const float bv = b2[e * ODIM + ocol];
      #pragma unroll
      for (int r = 0; r < 8; ++r) {
        const int ml = mrow + r + mhi;
        const float val = wts[ml] * (acc[t][r] + bv);
        atomicAdd(out + (size_t)toks[ml] * ODIM + ocol, val);
      }
    }
  }
}

// ---------------------------------------------------------------------------
extern "C" void kernel_launch(void* const* d_in, const int* in_sizes, int n_in,
                              void* d_out, int out_size, void* d_ws, size_t ws_size,
                              hipStream_t stream) {
  const float* x  = (const float*)d_in[0];
  const float* Wg = (const float*)d_in[1];
  const float* bg = (const float*)d_in[2];
  const float* W1 = (const float*)d_in[3];
  const float* b1 = (const float*)d_in[4];
  const float* W2 = (const float*)d_in[5];
  const float* b2 = (const float*)d_in[6];
  float* out = (float*)d_out;

  // Workspace layout: counts[8] | tok_list[8*8192] | w_list[8*8192]  (~512 KB)
  int*   counts   = (int*)d_ws;
  int*   tok_list = counts + NEXP;
  float* w_list   = (float*)(tok_list + NEXP * NTOK);

  hipMemsetAsync(counts, 0, NEXP * sizeof(int), stream);
  hipMemsetAsync(out, 0, (size_t)NTOK * ODIM * sizeof(float), stream);

  moe_gate<<<NTOK / 8, 256, 0, stream>>>(x, Wg, bg, counts, tok_list, w_list);

  const size_t lds =
      (size_t)(BM * XS_STRIDE + HC * W1_STRIDE + BM * HS_STRIDE + BO * W2_STRIDE) * 2
      + BM * sizeof(int) + BM * sizeof(float);
  dim3 grid(NTOK / BM, ODIM / BO, NEXP);  // worst-case M tiles; blocks early-exit on counts
  moe_expert_ffn<<<grid, 256, lds, stream>>>(x, W1, b1, W2, b2,
                                             counts, tok_list, w_list, out);
}
